// Module_87342454931937
// MI455X (gfx1250) — matrix operand order repaired and checked by reading, not compiled, not run on hardware
//
#include <hip/hip_runtime.h>
#include <hip/hip_bf16.h>

// ---------------------------------------------------------------------------
// Problem constants (match reference)
// ---------------------------------------------------------------------------
#define D_DIM   1024
#define L_DIM   2048
#define B_DIM   32
#define ROWS    (B_DIM * L_DIM)          // 65536
#define MT      32                       // rows per workgroup
#define NT      512                      // cols per workgroup (2 n-blocks)
#define KC      32                       // K chunk per WMMA step
#define NCH     (D_DIM / KC)             // 32 chunks
#define BPITCH  80                       // padded B-tile row pitch (bytes)
#define SIGMA_F 0.1f
#define HALF_SIG2 0.005f
#define LN_EPS_F 1e-5f

typedef __attribute__((ext_vector_type(16))) __bf16 v16bf;
typedef __attribute__((ext_vector_type(8)))  float  v8f;
typedef __attribute__((ext_vector_type(4)))  int    v4i;

enum { MODE_QP = 0, MODE_KP = 1, MODE_VP = 2, MODE_PRIOR = 3 };

// ---------------------------------------------------------------------------
// CDNA5 async global->LDS copy (ASYNCcnt path), guarded so the file always
// compiles; fallback is a plain LDS store (synchronous).
// ---------------------------------------------------------------------------
#if __has_builtin(__builtin_amdgcn_global_load_async_to_lds_b128)
#define ASYNC_OK 1
#else
#define ASYNC_OK 0
#endif

__device__ __forceinline__ void async_copy16(const void* g, void* l)
{
#if ASYNC_OK
    __builtin_amdgcn_global_load_async_to_lds_b128(
        (__attribute__((address_space(1))) v4i*)g,
        (__attribute__((address_space(3))) v4i*)l, 0, 0);
#else
    *(uint4*)l = *(const uint4*)g;
#endif
}

template <int N>
__device__ __forceinline__ void wait_async()
{
#if ASYNC_OK
#if __has_builtin(__builtin_amdgcn_s_wait_asynccnt)
    __builtin_amdgcn_s_wait_asynccnt(N);
#else
    asm volatile("s_wait_asynccnt %0" ::"i"(N) : "memory");
#endif
#endif
}

// ---------------------------------------------------------------------------
// f32 -> bf16 conversion (weights): tiny one-time pass so B staging can use
// raw async byte copies.
// ---------------------------------------------------------------------------
__global__ __launch_bounds__(256) void cvt_bf16_kernel(
    const float* __restrict__ s, __bf16* __restrict__ d, int n)
{
    const int i4 = (blockIdx.x * 256 + threadIdx.x) * 4;
    if (i4 + 3 < n) {
        const float4 v = *(const float4*)(s + i4);
        d[i4 + 0] = (__bf16)v.x;
        d[i4 + 1] = (__bf16)v.y;
        d[i4 + 2] = (__bf16)v.z;
        d[i4 + 3] = (__bf16)v.w;
    }
}

// ---------------------------------------------------------------------------
// Fused WMMA GEMM:  C[r, n] = A[r, :] @ W[n, :] + bias[n]
//   MODE_QP    : A f32 (Q, 32 rows)   -> store C f32 (Qp)
//   MODE_KP    : A f32 (K)            -> store C bf16 (Kp), dot with Qp[b]
//   MODE_VP    : A f32 (V)            -> store C bf16 (Vp)
//   MODE_PRIOR : A bf16 (Kp), relu(C) -> dot with phi2_w, no C store
// Double-buffered LDS; B tiles streamed with async global->LDS b128 copies.
// ---------------------------------------------------------------------------
template <int MODE>
__global__ __launch_bounds__(256) void gemm_wmma(
    const void* __restrict__ Ap,
    const __bf16* __restrict__ Wb,      // pre-converted bf16 weights [N,K]
    const float* __restrict__ bias,
    const float* __restrict__ dotvec,   // Qp (KP) or phi2_w (PRIOR)
    float* __restrict__ dotpart,        // [2][ROWS] partials
    __bf16* __restrict__ Cbf,
    float* __restrict__ Cf)
{
    // LDS: double-buffered tiles
    __shared__ v16bf aT[2][2][32];                        // 4 KB  (frag layout)
    __shared__ __align__(32) unsigned char bT[2][NT * BPITCH]; // 80 KB (row-major)
    __shared__ float sdot[2][4][16];

    const int tid  = threadIdx.x;
    const int lane = tid & 31;
    const int wid  = tid >> 5;
    const int wm   = wid >> 2;            // 0..1 : 16-row half
    const int wn   = wid & 3;             // 0..3 : 128-col strip
    const int rowBase = blockIdx.x * MT;
    const int nblk    = blockIdx.y;       // 0..1 : 512-col half
    const int cBase   = nblk * NT + wn * 128;

    const float*  Af = (const float*)Ap;
    const __bf16* Ab = (const __bf16*)Ap;
    const __bf16* W0 = Wb + (long)nblk * NT * D_DIM;   // row 0 of this n-block

    v8f acc[8] = {};

    // ---- staging helpers (inlined lambdas) --------------------------------
    auto issueB = [&](int buf, int kc) {
        unsigned char* bb = &bT[buf][0];
#pragma unroll
        for (int j = 0; j < 8; ++j) {               // 8 async b128 per wave
            const int seg  = ((wid * 8 + j) << 5) | lane;  // 0..2047
            const int nloc = seg >> 2;              // B-tile row (0..511)
            const int ks   = seg & 3;               // 16B segment in 64B row
            const __bf16* g = W0 + (long)nloc * D_DIM + kc + ks * 8;
            async_copy16(g, bb + nloc * BPITCH + ks * 16);
        }
    };
    auto stageA = [&](int buf, int kc) {
        if (tid < 64) {
            const int wmT = tid >> 5;
            const int le  = tid & 31;
            const int h   = le >> 4;
            const int m   = rowBase + wmT * 16 + (le & 15);
            const long rb = (long)m * D_DIM + kc + 8 * h;
            __bf16* dst   = (__bf16*)&aT[buf][wmT][le];
#pragma unroll
            for (int i = 0; i < 8; ++i) {
                if (MODE == MODE_PRIOR) {
                    dst[i]     = Ab[rb + i];
                    dst[i + 8] = Ab[rb + 16 + i];
                } else {
                    dst[i]     = (__bf16)Af[rb + i];
                    dst[i + 8] = (__bf16)Af[rb + 16 + i];
                }
            }
        }
    };

    // ---- software pipeline ------------------------------------------------
    issueB(0, 0);
    stageA(0, 0);

    for (int i = 0; i < NCH; ++i) {
        const int cur = i & 1;
        if (i + 1 < NCH) {
            issueB(1 - cur, (i + 1) * KC);   // prefetch next chunk
            stageA(1 - cur, (i + 1) * KC);
            wait_async<8>();                 // chunk i resident, i+1 in flight
        } else {
            wait_async<0>();
        }
        __syncthreads();

        // load all fragments, then issue WMMAs back-to-back
        const v16bf aF = aT[cur][wm][lane];
        const int h  = lane >> 4;
        const int nl = lane & 15;
        union FragU { v16bf v; uint4 q[2]; };
        v16bf bF[8];
#pragma unroll
        for (int s = 0; s < 8; ++s) {
            const int nloc = (wn * 8 + s) * 16 + nl;
            const unsigned char* p = &bT[cur][nloc * BPITCH + h * 16];
            FragU u;
            u.q[0] = *(const uint4*)p;
            u.q[1] = *(const uint4*)(p + 32);
            bF[s] = u.v;
        }
#pragma unroll
        for (int s = 0; s < 8; ++s)
            acc[s] = __builtin_amdgcn_wmma_f32_16x16x32_bf16(
                false, aF, false, bF[s], (short)0, acc[s], false, false);

        __syncthreads();
    }

    // ------------------------------ epilogue -------------------------------
    const int h    = lane >> 4;
    const int nl   = lane & 15;
    const int row0 = rowBase + wm * 16;
    const int b    = row0 >> 11;          // batch (tiles never cross batches)

    float pm[8];
#pragma unroll
    for (int v = 0; v < 8; ++v) pm[v] = 0.0f;

#pragma unroll
    for (int s = 0; s < 8; ++s) {
        const int n    = cBase + s * 16 + nl;
        const float bv = bias[n];
        float dv = 0.0f;
        if (MODE == MODE_KP)    dv = dotvec[b * D_DIM + n];
        if (MODE == MODE_PRIOR) dv = dotvec[n];
#pragma unroll
        for (int v = 0; v < 8; ++v) {
            float c = acc[s][v] + bv;
            const int  m   = v + 8 * h;
            const long idx = (long)(row0 + m) * D_DIM + n;
            if (MODE == MODE_QP)  Cf[idx] = c;
            if (MODE == MODE_VP)  Cbf[idx] = (__bf16)c;
            if (MODE == MODE_KP) { Cbf[idx] = (__bf16)c; pm[v] += c * dv; }
            if (MODE == MODE_PRIOR) { c = fmaxf(c, 0.0f); pm[v] += c * dv; }
        }
    }

    if (MODE == MODE_KP || MODE == MODE_PRIOR) {
#pragma unroll
        for (int v = 0; v < 8; ++v) {
#pragma unroll
            for (int mask = 1; mask < 16; mask <<= 1)
                pm[v] += __shfl_xor(pm[v], mask, 32);
        }
        if (nl == 0) {
#pragma unroll
            for (int v = 0; v < 8; ++v)
                sdot[wm][wn][v + 8 * h] = pm[v];
        }
        __syncthreads();
        if (tid < 32) {
            const int tm = tid >> 4, mm = tid & 15;
            const float s = ((sdot[tm][0][mm] + sdot[tm][1][mm]) +
                             (sdot[tm][2][mm] + sdot[tm][3][mm]));
            dotpart[(long)nblk * ROWS + rowBase + tid] = s;
        }
    }
}

// ---------------------------------------------------------------------------
// prior_mu[r] = prp[r] + prp[ROWS+r] + phi2_b - 0.5*sigma^2
// ---------------------------------------------------------------------------
__global__ void prior_combine_kernel(const float* __restrict__ prp,
                                     const float* __restrict__ phi2_b,
                                     float* __restrict__ out_prior)
{
    const int i = blockIdx.x * blockDim.x + threadIdx.x;
    if (i < ROWS)
        out_prior[i] = prp[i] + prp[ROWS + i] + phi2_b[0] - HALF_SIG2;
}

// ---------------------------------------------------------------------------
// Per-batch: post_mu -> log-normal sample -> softplus simplex weights,
// ctx = weights @ Vp, LayerNorm, +Qp residual.
// ---------------------------------------------------------------------------
__global__ __launch_bounds__(256) void finalize_kernel(
    const float* __restrict__ phip,
    const float* __restrict__ eps,
    const int*  __restrict__ padding,
    const __bf16* __restrict__ Vp,
    const float* __restrict__ Qp,
    const float* __restrict__ ln_g,
    const float* __restrict__ ln_b,
    float* __restrict__ out_ctx,
    float* __restrict__ out_w)
{
    __shared__ float wsh[L_DIM];
    __shared__ float red[256];

    const int b   = blockIdx.x;
    const int tid = threadIdx.x;

    float lsum = 0.0f;
    for (int l = tid; l < L_DIM; l += 256) {
        const int gi = b * L_DIM + l;
        float sp;
        if (padding[gi]) {
            sp = 0.0f;                                 // softplus(-inf) = 0
        } else {
            const float phi     = (phip[gi] + phip[ROWS + gi]) * (1.0f / 32.0f);
            const float post_mu = phi - HALF_SIG2;
            const float sample  = __expf(post_mu + SIGMA_F * eps[gi]);
            sp = (sample > 20.0f) ? sample : log1pf(__expf(sample));
        }
        wsh[l] = sp;
        lsum += sp;
    }
    red[tid] = lsum;
    __syncthreads();
    for (int s = 128; s > 0; s >>= 1) {
        if (tid < s) red[tid] += red[tid + s];
        __syncthreads();
    }
    const float inv = 1.0f / red[0];
    for (int l = tid; l < L_DIM; l += 256) {
        const float w = wsh[l] * inv;
        wsh[l] = w;
        out_w[b * L_DIM + l] = w;
    }
    __syncthreads();

    float ctx[4] = {0.0f, 0.0f, 0.0f, 0.0f};
    const __bf16* vb = Vp + (long)b * L_DIM * D_DIM;
    for (int l = 0; l < L_DIM; ++l) {
        const float wl = wsh[l];
        const __bf16* row = vb + (long)l * D_DIM;
#pragma unroll
        for (int j = 0; j < 4; ++j)
            ctx[j] += wl * (float)row[tid + j * 256];
    }

    const float s1 = (ctx[0] + ctx[1]) + (ctx[2] + ctx[3]);
    const float s2 = (ctx[0] * ctx[0] + ctx[1] * ctx[1]) +
                     (ctx[2] * ctx[2] + ctx[3] * ctx[3]);
    red[tid] = s1;
    __syncthreads();
    for (int s = 128; s > 0; s >>= 1) {
        if (tid < s) red[tid] += red[tid + s];
        __syncthreads();
    }
    const float mean = red[0] * (1.0f / (float)D_DIM);
    __syncthreads();
    red[tid] = s2;
    __syncthreads();
    for (int s = 128; s > 0; s >>= 1) {
        if (tid < s) red[tid] += red[tid + s];
        __syncthreads();
    }
    const float var  = red[0] * (1.0f / (float)D_DIM) - mean * mean;
    const float rstd = rsqrtf(var + LN_EPS_F);

#pragma unroll
    for (int j = 0; j < 4; ++j) {
        const int c = tid + j * 256;
        out_ctx[b * D_DIM + c] =
            (ctx[j] - mean) * rstd * ln_g[c] + ln_b[c] + Qp[b * D_DIM + c];
    }
}

// ---------------------------------------------------------------------------
extern "C" void kernel_launch(void* const* d_in, const int* in_sizes, int n_in,
                              void* d_out, int out_size, void* d_ws, size_t ws_size,
                              hipStream_t stream)
{
    (void)in_sizes; (void)n_in; (void)out_size; (void)ws_size;

    const float* Q   = (const float*)d_in[0];
    const float* K   = (const float*)d_in[1];
    const float* V   = (const float*)d_in[2];
    const float* eps = (const float*)d_in[3];
    const float* Wq  = (const float*)d_in[4];
    const float* bq  = (const float*)d_in[5];
    const float* Wk  = (const float*)d_in[6];
    const float* bk  = (const float*)d_in[7];
    const float* Wv  = (const float*)d_in[8];
    const float* bv  = (const float*)d_in[9];
    const float* p1w = (const float*)d_in[10];
    const float* p1b = (const float*)d_in[11];
    const float* p2w = (const float*)d_in[12];
    const float* p2b = (const float*)d_in[13];
    const float* lng = (const float*)d_in[14];
    const float* lnb = (const float*)d_in[15];
    const int*   pad = (const int*)d_in[16];

    float* out_ctx   = (float*)d_out;                 // [B, D]
    float* out_w     = out_ctx + B_DIM * D_DIM;       // [B, L]
    float* out_prior = out_w + B_DIM * L_DIM;         // [B, L]

    char* ws = (char*)d_ws;
    float*  qp   = (float*)ws;  ws += (size_t)B_DIM * D_DIM * sizeof(float);
    float*  phip = (float*)ws;  ws += (size_t)2 * ROWS * sizeof(float);
    float*  prp  = (float*)ws;  ws += (size_t)2 * ROWS * sizeof(float);
    __bf16* wqb  = (__bf16*)ws; ws += (size_t)D_DIM * D_DIM * sizeof(__bf16);
    __bf16* wkb  = (__bf16*)ws; ws += (size_t)D_DIM * D_DIM * sizeof(__bf16);
    __bf16* wvb  = (__bf16*)ws; ws += (size_t)D_DIM * D_DIM * sizeof(__bf16);
    __bf16* p1wb = (__bf16*)ws; ws += (size_t)D_DIM * D_DIM * sizeof(__bf16);
    __bf16* kp   = (__bf16*)ws; ws += (size_t)ROWS * D_DIM * sizeof(__bf16);
    __bf16* vp   = (__bf16*)ws; ws += (size_t)ROWS * D_DIM * sizeof(__bf16);

    const int NW = D_DIM * D_DIM;          // 1M elements per weight
    const int cvtGrid = NW / (256 * 4);

    // 0) weights -> bf16 (enables raw async B staging)
    cvt_bf16_kernel<<<cvtGrid, 256, 0, stream>>>(Wq,  wqb,  NW);
    cvt_bf16_kernel<<<cvtGrid, 256, 0, stream>>>(Wk,  wkb,  NW);
    cvt_bf16_kernel<<<cvtGrid, 256, 0, stream>>>(Wv,  wvb,  NW);
    cvt_bf16_kernel<<<cvtGrid, 256, 0, stream>>>(p1w, p1wb, NW);

    // 1) Qp = Q @ Wq^T + bq  (f32, small)
    gemm_wmma<MODE_QP><<<dim3(1, 2), 256, 0, stream>>>(
        Q, wqb, bq, nullptr, nullptr, nullptr, qp);

    // 2) Kp = K @ Wk^T + bk  (bf16) fused with phi = Qp . Kp partials
    gemm_wmma<MODE_KP><<<dim3(ROWS / MT, 2), 256, 0, stream>>>(
        K, wkb, bk, qp, phip, kp, nullptr);

    // 3) Vp = V @ Wv^T + bv  (bf16)
    gemm_wmma<MODE_VP><<<dim3(ROWS / MT, 2), 256, 0, stream>>>(
        V, wvb, bv, nullptr, nullptr, vp, nullptr);

    // 4) prior: relu(Kp @ phi1_w^T + b) . phi2_w  (h never materialized)
    gemm_wmma<MODE_PRIOR><<<dim3(ROWS / MT, 2), 256, 0, stream>>>(
        kp, p1wb, p1b, p2w, prp, nullptr, nullptr);

    // 5) prior_mu = partials + phi2_b - 0.5*sigma^2
    prior_combine_kernel<<<ROWS / 256, 256, 0, stream>>>(prp, p2b, out_prior);

    // 6) weights / ctx / layernorm / residual
    finalize_kernel<<<B_DIM, 256, 0, stream>>>(
        phip, eps, pad, vp, qp, lng, lnb, out_ctx, out_w);
}
